// Linear_Attention_68564857914100
// MI455X (gfx1250) — matrix operand order repaired and checked by reading, not compiled, not run on hardware
//
#include <hip/hip_runtime.h>
#include <hip/hip_bf16.h>

// ---------------------------------------------------------------------------
// CDNA5 (gfx1250) linear-attention pipeline, bf16 WMMA everywhere.
//   B=4, S=2048, D=1024.
// All six GEMM stages run through one generic v_wmma_f32_16x16x32_bf16 kernel
// with double-buffered LDS staging; uses CDNA5 async global->LDS copies
// (ASYNCcnt) when the toolchain exposes the builtin, else load+ds_store.
// ---------------------------------------------------------------------------

typedef __attribute__((ext_vector_type(16))) __bf16 v16bf;
typedef __attribute__((ext_vector_type(8)))  float  v8f;
typedef __attribute__((ext_vector_type(4)))  int    v4i;

#define BB 4
#define SS 2048
#define DD 1024

#if defined(__AMDGCN__) && \
    __has_builtin(__builtin_amdgcn_global_load_async_to_lds_b128) && \
    __has_builtin(__builtin_amdgcn_s_wait_asynccnt)
#define USE_ASYNC_LDS 1
typedef __attribute__((address_space(1))) v4i gv4i;   // global int4
typedef __attribute__((address_space(3))) v4i sv4i;   // LDS int4
#else
#define USE_ASYNC_LDS 0
#endif

// ---- helpers --------------------------------------------------------------

__device__ __forceinline__ unsigned short f2bf(float f) {
    union { float f; unsigned int u; } v; v.f = f;
    unsigned int u = v.u;
    u += 0x7FFFu + ((u >> 16) & 1u);   // round-to-nearest-even
    return (unsigned short)(u >> 16);
}

__device__ __forceinline__ void copy16_to_lds(unsigned short* dst,
                                              const unsigned short* src) {
#if USE_ASYNC_LDS
    __builtin_amdgcn_global_load_async_to_lds_b128(
        (gv4i*)src, (sv4i*)dst, 0, 0);
#else
    *(uint4*)dst = *(const uint4*)src;
#endif
}

// ---- conversion kernels ---------------------------------------------------

__global__ __launch_bounds__(256) void cvt_f32_bf16(const float* __restrict__ in,
                                                    unsigned short* __restrict__ out,
                                                    long long n) {
    long long i = (long long)blockIdx.x * blockDim.x + threadIdx.x;
    long long stride = (long long)gridDim.x * blockDim.x;
    for (; i < n; i += stride) out[i] = f2bf(in[i]);
}

// Wt[e*D + d] = bf16(W[d*D + e])   (store W^T so GEMM B-operand is N x K row-major)
__global__ __launch_bounds__(256) void transpose_f32_bf16(const float* __restrict__ W,
                                                          unsigned short* __restrict__ Wt,
                                                          int D) {
    int i = blockIdx.x * blockDim.x + threadIdx.x;
    if (i < D * D) {
        int e = i / D, d = i % D;
        Wt[i] = f2bf(W[(size_t)d * D + e]);
    }
}

// ---- generic bf16 WMMA GEMM ----------------------------------------------
// C[M,N] = A[M,K] * Bt[N,K]^T (+bias) (optional cos)
//   A  : bf16 row-major, lda = K
//   Bt : bf16 row-major N x K with row stride ldb (columns of B are rows of Bt)
// Outputs (any non-null): outF (f32 row-major, ld=N), outB (bf16 row-major),
//   outBT (bf16 transposed: outBT[n*M + m]).
// Tile: 128x64 per workgroup, BK=32, 256 threads = 8 waves,
//   wave w -> rows [16w,16w+16), 4 accumulators covering 64 cols.
// LDS double-buffered: prefetch tile k+1 overlaps the 4 WMMAs of tile k.

#define BM 128
#define BN 64
#define BK 32

__device__ __forceinline__ void stage_tile(unsigned short* Abuf,
                                           unsigned short* Bbuf,
                                           const unsigned short* __restrict__ A,
                                           const unsigned short* __restrict__ Bt,
                                           int m0, int n0, int k0,
                                           int K, int ldb, int tid) {
    // A tile 128x32 bf16 = 512 x 16B, B tile 64x32 = 256 x 16B; 256 threads.
    #pragma unroll
    for (int i = 0; i < 2; ++i) {
        int idx = tid + i * 256;                 // 0..511
        int row = idx >> 2, ch = idx & 3;
        copy16_to_lds(&Abuf[row * BK + ch * 8],
                      &A[(size_t)(m0 + row) * K + k0 + ch * 8]);
    }
    {
        int row = tid >> 2, ch = tid & 3;
        copy16_to_lds(&Bbuf[row * BK + ch * 8],
                      &Bt[(size_t)(n0 + row) * ldb + k0 + ch * 8]);
    }
}

__global__ __launch_bounds__(256) void gemm_bf16_wmma(
    const unsigned short* __restrict__ A,
    const unsigned short* __restrict__ Bt, int ldb,
    const float* __restrict__ bias,
    int M, int N, int K, int act,
    float* __restrict__ outF,
    unsigned short* __restrict__ outB,
    unsigned short* __restrict__ outBT) {

    __shared__ __align__(16) unsigned short Atile[2][BM * BK];   // 2 x 8 KB
    __shared__ __align__(16) unsigned short Btile[2][BN * BK];   // 2 x 4 KB

    const int m0   = blockIdx.x * BM;
    const int n0   = blockIdx.y * BN;
    const int tid  = threadIdx.x;
    const int lane = tid & 31;
    const int w    = tid >> 5;            // wave id 0..7
    const int h    = lane >> 4;           // half-wave 0/1
    const int rr   = lane & 15;

    v8f acc[4];
    #pragma unroll
    for (int i = 0; i < 4; ++i) acc[i] = (v8f){0.f,0.f,0.f,0.f,0.f,0.f,0.f,0.f};

    union Frag { v16bf v; uint4 q[2]; };

    stage_tile(Atile[0], Btile[0], A, Bt, m0, n0, 0, K, ldb, tid);

    for (int k0 = 0; k0 < K; k0 += BK) {
        const int cur = (k0 / BK) & 1;
#if USE_ASYNC_LDS
        __builtin_amdgcn_s_wait_asynccnt(0);   // own prefetch landed in LDS
#endif
        __syncthreads();                       // everyone's prefetch + prev compute done
        if (k0 + BK < K)
            stage_tile(Atile[cur ^ 1], Btile[cur ^ 1], A, Bt,
                       m0, n0, k0 + BK, K, ldb, tid);

        // A fragment: lane half 0 -> K 0..7 & 16..23 ; half 1 -> K 8..15 & 24..31
        Frag af;
        {
            int arow = w * 16 + rr;
            af.q[0] = *(const uint4*)&Atile[cur][arow * BK + h * 8];
            af.q[1] = *(const uint4*)&Atile[cur][arow * BK + h * 8 + 16];
        }
        #pragma unroll
        for (int ct = 0; ct < 4; ++ct) {
            // B fragment: lane half 0 -> K 0..7 & 16..23 of col n ; half 1 -> K 8..15 & 24..31
            Frag bfr;
            int bn = ct * 16 + rr;
            bfr.q[0] = *(const uint4*)&Btile[cur][bn * BK + h * 8];
            bfr.q[1] = *(const uint4*)&Btile[cur][bn * BK + h * 8 + 16];
            acc[ct] = __builtin_amdgcn_wmma_f32_16x16x32_bf16(
                false, af.v, false, bfr.v, (short)0, acc[ct], false, false);
        }
    }

    // epilogue: C layout -> VGPR r, lanes 0-15: M=r, lanes 16-31: M=8+r; N = lane&15
    const int hm = h * 8;
    #pragma unroll
    for (int ct = 0; ct < 4; ++ct) {
        #pragma unroll
        for (int r = 0; r < 8; ++r) {
            int row = m0 + w * 16 + hm + r;
            int col = n0 + ct * 16 + rr;
            float v = acc[ct][r];
            if (bias) v += bias[col];
            if (act)  v = cosf(v);
            if (outF)  outF[(size_t)row * N + col]  = v;
            if (outB)  outB[(size_t)row * N + col]  = f2bf(v);
            if (outBT) outBT[(size_t)col * M + row] = f2bf(v);
        }
    }
}

// ---- row softmax (wave32 per row), f32 scores -> bf16 probs --------------

__global__ __launch_bounds__(256) void softmax_rows(const float* __restrict__ Sc,
                                                    unsigned short* __restrict__ P,
                                                    int rows, int cols) {
    int r    = blockIdx.x * 8 + (threadIdx.x >> 5);
    int lane = threadIdx.x & 31;
    if (r >= rows) return;
    const float* sr = Sc + (size_t)r * cols;
    unsigned short* pr = P + (size_t)r * cols;

    float m = -3.402823466e+38f;
    for (int k = lane; k < cols; k += 32) m = fmaxf(m, sr[k]);
    #pragma unroll
    for (int o = 16; o > 0; o >>= 1) m = fmaxf(m, __shfl_xor(m, o));

    float s = 0.f;
    for (int k = lane; k < cols; k += 32) s += __expf(sr[k] - m);
    #pragma unroll
    for (int o = 16; o > 0; o >>= 1) s += __shfl_xor(s, o);

    float inv = 1.0f / s;
    for (int k = lane; k < cols; k += 32) pr[k] = f2bf(__expf(sr[k] - m) * inv);
}

// ---- driver ---------------------------------------------------------------

extern "C" void kernel_launch(void* const* d_in, const int* in_sizes, int n_in,
                              void* d_out, int out_size, void* d_ws, size_t ws_size,
                              hipStream_t stream) {
    const float* x  = (const float*)d_in[0];
    const float* y  = (const float*)d_in[1];
    const float* Wq = (const float*)d_in[2];
    const float* bq = (const float*)d_in[3];
    const float* Wk = (const float*)d_in[4];
    const float* bk = (const float*)d_in[5];
    const float* Wv = (const float*)d_in[6];
    const float* bv = (const float*)d_in[7];
    const float* Wr = (const float*)d_in[8];
    const float* br = (const float*)d_in[9];
    float* out = (float*)d_out;

    const size_t nBSD = (size_t)BB * SS * DD;   // 8,388,608
    const size_t nDD  = (size_t)DD * DD;        // 1,048,576
    const size_t nBSS = (size_t)BB * SS * SS;   // 16,777,216

    unsigned short* xb   = (unsigned short*)d_ws;
    unsigned short* yb   = xb   + nBSD;
    unsigned short* WqT  = yb   + nBSD;
    unsigned short* WkT  = WqT  + nDD;
    unsigned short* WvT  = WkT  + nDD;
    unsigned short* WrT  = WvT  + nDD;
    unsigned short* Qb   = WrT  + nDD;
    unsigned short* Kb   = Qb   + nBSD;
    unsigned short* Khb  = Kb   + nBSD;
    unsigned short* Vtb  = Khb  + nBSD;          // [d][b*S + s], row stride B*S
    float*          scor = (float*)(Vtb + nBSD);
    unsigned short* prob = (unsigned short*)(scor + nBSS);
    size_t need = (size_t)((char*)(prob + nBSS) - (char*)d_ws);
    if (ws_size < need) return;   // workspace too small; avoid OOB

    const int Mtot = BB * SS;     // 8192
    dim3 blk(256);

    // Stage 0: convert activations + weights to bf16 (weights transposed)
    cvt_f32_bf16<<<2048, blk, 0, stream>>>(x, xb, (long long)nBSD);
    cvt_f32_bf16<<<2048, blk, 0, stream>>>(y, yb, (long long)nBSD);
    int trBlocks = (int)(nDD / 256);
    transpose_f32_bf16<<<trBlocks, blk, 0, stream>>>(Wq, WqT, DD);
    transpose_f32_bf16<<<trBlocks, blk, 0, stream>>>(Wk, WkT, DD);
    transpose_f32_bf16<<<trBlocks, blk, 0, stream>>>(Wv, WvT, DD);
    transpose_f32_bf16<<<trBlocks, blk, 0, stream>>>(Wr, WrT, DD);

    // Stage 1: projections (merged batch, M = B*S)
    dim3 gP(Mtot / BM, DD / BN);  // 64 x 16
    gemm_bf16_wmma<<<gP, blk, 0, stream>>>(xb, WqT, DD, bq, Mtot, DD, DD, 0,
                                           nullptr, Qb, nullptr);
    gemm_bf16_wmma<<<gP, blk, 0, stream>>>(yb, WkT, DD, bk, Mtot, DD, DD, 0,
                                           nullptr, Kb, nullptr);
    gemm_bf16_wmma<<<gP, blk, 0, stream>>>(yb, WvT, DD, bv, Mtot, DD, DD, 0,
                                           nullptr, nullptr, Vtb);   // V^T bf16
    // K_hat = cos(K @ Wr + br)
    gemm_bf16_wmma<<<gP, blk, 0, stream>>>(Kb, WrT, DD, br, Mtot, DD, DD, 1,
                                           nullptr, Khb, nullptr);

    // Stage 2: scores = Q @ Khat^T  (Khat row-major == N x K operand)
    dim3 gS(SS / BM, SS / BN);    // 16 x 32
    for (int b = 0; b < BB; ++b) {
        gemm_bf16_wmma<<<gS, blk, 0, stream>>>(Qb + (size_t)b * SS * DD,
                                               Khb + (size_t)b * SS * DD, DD,
                                               nullptr, SS, SS, DD, 0,
                                               scor + (size_t)b * SS * SS,
                                               nullptr, nullptr);
    }

    // Stage 3: softmax rows -> bf16 probs
    softmax_rows<<<(BB * SS) / 8, blk, 0, stream>>>(scor, prob, BB * SS, SS);

    // Stage 4: out = probs @ V   (B-operand = V^T rows, stride B*S)
    dim3 gO(SS / BM, DD / BN);    // 16 x 16
    for (int b = 0; b < BB; ++b) {
        gemm_bf16_wmma<<<gO, blk, 0, stream>>>(prob + (size_t)b * SS * SS,
                                               Vtb + (size_t)b * SS, Mtot,
                                               nullptr, SS, DD, SS, 0,
                                               out + (size_t)b * SS * DD,
                                               nullptr, nullptr);
    }
}
